// ReconstructionOriginalLayer_28346784153740
// MI455X (gfx1250) — matrix operand order repaired and checked by reading, not compiled, not run on hardware
//
#include <hip/hip_runtime.h>

// ============================================================================
// MRI SENSE recon via CG; fft2c/ifft2c as DFT matmuls (W·X·W) on CDNA5 WMMA.
//
// Round-3 GEMM data flow (fully async staging, double buffered):
//  - W is symmetric, and pass-1 GEMMs (t = X*W) store t TRANSPOSED. Hence the
//    B operand of *every* GEMM is available transposed in global memory
//    (W itself, or t^T), so A-tiles AND B-tiles are row-major byte copies:
//    all staging uses global_load_async_to_lds_b128 (ASYNCcnt), double
//    buffered so tile k+1 copies overlap tile k's WMMA work.
//  - Transposed epilogue goes through LDS (scatter in LDS, coalesced 16B
//    global stores), overlaid on the staging pool after the K-loop.
//  - WMMA: v_wmma_f32_16x16x32_f16, f32 accumulation; 4 real MMAs per complex
//    16x16 subtile (A-imag negated by VALU; f16 WMMA has C-negate only).
//  - Binary k-space mask fused into fwd-FFT pass-2 epilogue.
//
// Shapes fixed by setup_inputs(): B=1, S=4, C=16, H=W=320.  ws need ~67 MB.
// ============================================================================

typedef __attribute__((ext_vector_type(16))) _Float16 v16h;
typedef __attribute__((ext_vector_type(8)))  float    v8f;

constexpr int   N320 = 320;
constexpr int   HW   = N320 * N320;
constexpr long  PL2  = 2L * HW;       // halfs per complex plane-pair image
constexpr int   S_   = 4;
constexpr int   C_   = 16;
constexpr int   SC   = S_ * C_;
constexpr float RHO  = 0.1f;
constexpr float EPS_ = 1e-12f;
constexpr int   CG_ITERS = 15;
constexpr int   NBLK = 240;

// ---------------------------------------------------------------------------
// DFT matrices as f16 planes: W[j,k] = (1/sqrt(320)) e^{-2pi i (j+160)(k+160)/320}
// (fftshift folded in; W symmetric => fft2c(X) = W X W; ifft uses conj)
// ---------------------------------------------------------------------------
__global__ void build_dft_f16(_Float16* __restrict__ Wf, _Float16* __restrict__ Wi) {
    int idx = blockIdx.x * 256 + threadIdx.x;
    if (idx >= HW) return;
    int j = idx / N320, k = idx % N320;
    int t = ((j + 160) * (k + 160)) % N320;
    float ang = 6.283185307179586f * (float)t / 320.0f;
    float sn = __sinf(ang), cs = __cosf(ang);
    const float sc = 0.05590169943749474f;          // 1/sqrt(320), ortho
    Wf[idx]      = (_Float16)(cs * sc);
    Wf[HW + idx] = (_Float16)(-sn * sc);
    Wi[idx]      = (_Float16)(cs * sc);
    Wi[HW + idx] = (_Float16)(sn * sc);
}

// ---------------------------------------------------------------------------
// Batched complex GEMM on f16 planes: C[b] = A[b?] * B[b?] where the B operand
// is supplied TRANSPOSED (BT).  320x320 complex, 64x64 WG tile, 4 waves,
// 32x32 complex per wave (2x2 WMMA subtiles), K stepped by 32.
// ---------------------------------------------------------------------------
#define TM  64
#define TN  64
#define KB  32
#define KBP 40   // LDS tile row stride (halfs): 80B = 5*16B, B128-aligned rows
#define TMP 72   // LDS transpose-scratch row stride (halfs): 144B, 16B aligned

union FragH { v16h v; _Float16 h[16]; unsigned u[8]; };

// pool layout: [buf][A|B][plane][row<64][KBP]  (20480 halfs = 40KB);
// transpose scratch (2*64*TMP = 9216 halfs) overlays it after the K-loop.
__device__ __forceinline__ int stageOff(int buf, int ab, int pl) {
    return (((buf * 2) + ab) * 2 + pl) * (TM * KBP);
}

__device__ __forceinline__ void stage_async(const _Float16* __restrict__ G, int r0,
                                            int kk, _Float16* pool, int buf, int ab,
                                            int tid) {
    #pragma unroll
    for (int e = 0; e < 4; ++e) {                 // 512 x 16B chunks / 128 thr
        int idx   = e * 128 + tid;
        int plane = idx >> 8;
        int rem   = idx & 255;
        int row   = rem >> 2;
        int c16   = rem & 3;
        const _Float16* g = G + (long)plane * HW + (r0 + row) * N320 + kk + c16 * 8;
        unsigned lds = (unsigned)(uintptr_t)
            &pool[stageOff(buf, ab, plane) + row * KBP + c16 * 8];
        asm volatile("global_load_async_to_lds_b128 %0, %1, off"
                     :: "v"(lds), "v"(g) : "memory");
    }
}

__global__ __launch_bounds__(128)
void cgemm_wmma(const _Float16* __restrict__ Ap, const _Float16* __restrict__ BTp,
                _Float16* __restrict__ C16, float2* __restrict__ C32,
                const float* __restrict__ mask,
                long aB, long bB, long cB, int storeT)
{
    __shared__ __align__(16) _Float16 pool[2 * 2 * 2 * TM * KBP];

    const int tid  = threadIdx.x;
    const int lane = tid & 31;
    const int wave = tid >> 5;
    const int m0   = blockIdx.x * TM;
    const int n0   = blockIdx.y * TN;
    const long b   = blockIdx.z;

    const _Float16* Ag = Ap  + b * aB;   // real plane; +HW for imag
    const _Float16* Bg = BTp + b * bB;   // transposed B: rows are B columns

    const int wm = (wave >> 1) * 32;
    const int wn = (wave & 1) * 32;
    const int ml = lane & 15;
    const int hh = lane >> 4;

    v8f accR[2][2] = {};
    v8f accI[2][2] = {};

    // prologue: stage tile 0 into buffer 0
    stage_async(Ag, m0, 0, pool, 0, 0, tid);
    stage_async(Bg, n0, 0, pool, 0, 1, tid);
    asm volatile("s_wait_asynccnt 0x0" ::: "memory");
    __syncthreads();

    for (int kk = 0; kk < N320; kk += KB) {
        const int cur = (kk >> 5) & 1;
        if (kk + KB < N320) {                       // overlap: copy next tile
            stage_async(Ag, m0, kk + KB, pool, cur ^ 1, 0, tid);
            stage_async(Bg, n0, kk + KB, pool, cur ^ 1, 1, tid);
        }
        const _Float16* Ar = pool + stageOff(cur, 0, 0);
        const _Float16* Ai = pool + stageOff(cur, 0, 1);
        const _Float16* Br = pool + stageOff(cur, 1, 0);
        const _Float16* Bi = pool + stageOff(cur, 1, 1);

        // ---- fragments (A 16x32 / B 32x16, contiguous k = 16*hh + s in both) ----
        FragH fAr[2], fAi[2], fAiN[2], fBr[2], fBi[2];
        #pragma unroll
        for (int sm = 0; sm < 2; ++sm) {
            const _Float16* ar = Ar + (wm + sm * 16 + ml) * KBP;
            const _Float16* ai = Ai + (wm + sm * 16 + ml) * KBP;
            #pragma unroll
            for (int v = 0; v < 8; ++v) {
                int k0 = 16 * hh + 2 * v;
                fAr[sm].u[v] = *(const unsigned*)(ar + k0);
                fAi[sm].u[v] = *(const unsigned*)(ai + k0);
            }
            #pragma unroll
            for (int q = 0; q < 16; ++q) fAiN[sm].h[q] = -fAi[sm].h[q];
        }
        #pragma unroll
        for (int sn = 0; sn < 2; ++sn) {
            const _Float16* br = Br + (wn + sn * 16 + ml) * KBP;
            const _Float16* bi = Bi + (wn + sn * 16 + ml) * KBP;
            #pragma unroll
            for (int v = 0; v < 8; ++v) {
                int k0 = 16 * hh + 2 * v;
                fBr[sn].u[v] = *(const unsigned*)(br + k0);
                fBi[sn].u[v] = *(const unsigned*)(bi + k0);
            }
        }

        // ---- complex product: 4 real WMMAs per subtile ----
        #pragma unroll
        for (int sm = 0; sm < 2; ++sm)
            #pragma unroll
            for (int sn = 0; sn < 2; ++sn) {
                accR[sm][sn] = __builtin_amdgcn_wmma_f32_16x16x32_f16(
                    false, fAr[sm].v,  false, fBr[sn].v, (short)0, accR[sm][sn], false, false);
                accR[sm][sn] = __builtin_amdgcn_wmma_f32_16x16x32_f16(
                    false, fAiN[sm].v, false, fBi[sn].v, (short)0, accR[sm][sn], false, false);
                accI[sm][sn] = __builtin_amdgcn_wmma_f32_16x16x32_f16(
                    false, fAr[sm].v,  false, fBi[sn].v, (short)0, accI[sm][sn], false, false);
                accI[sm][sn] = __builtin_amdgcn_wmma_f32_16x16x32_f16(
                    false, fAi[sm].v,  false, fBr[sn].v, (short)0, accI[sm][sn], false, false);
            }

        asm volatile("s_wait_asynccnt 0x0" ::: "memory");  // next tile landed
        __syncthreads();
    }

    // ---- epilogue ----
    if (storeT) {
        // transposed f16-plane store via LDS (pool reused; all waves past loop)
        _Float16* Tls = pool;                               // [pl][n<64][TMP]
        #pragma unroll
        for (int sm = 0; sm < 2; ++sm)
            #pragma unroll
            for (int sn = 0; sn < 2; ++sn)
                #pragma unroll
                for (int v = 0; v < 8; ++v) {
                    int lm = wm + sm * 16 + 8 * hh + v;
                    int ln = wn + sn * 16 + ml;
                    Tls[ln * TMP + lm]              = (_Float16)accR[sm][sn][v];
                    Tls[(TN + ln) * TMP + lm]       = (_Float16)accI[sm][sn][v];
                }
        __syncthreads();
        int pl = tid >> 6, n = tid & 63;                    // one row per thread
        const uint4* src = (const uint4*)&Tls[(pl * TN + n) * TMP];
        uint4* dst = (uint4*)(C16 + b * cB + (long)pl * HW + (n0 + n) * N320 + m0);
        #pragma unroll
        for (int c = 0; c < 8; ++c) dst[c] = src[c];        // 128B coalesced row
    } else {
        const float* mrow = mask ? (mask + (b >> 4) * (long)HW) : nullptr;  // s=b/16
        #pragma unroll
        for (int sm = 0; sm < 2; ++sm)
            #pragma unroll
            for (int sn = 0; sn < 2; ++sn)
                #pragma unroll
                for (int v = 0; v < 8; ++v) {
                    int gm = m0 + wm + sm * 16 + 8 * hh + v;
                    int gn = n0 + wn + sn * 16 + ml;
                    float re = accR[sm][sn][v];
                    float im = accI[sm][sn][v];
                    if (mrow) { float mm = mrow[gm * N320 + gn]; re *= mm; im *= mm; }
                    if (C16) {
                        _Float16* o = C16 + b * cB;
                        o[gm * N320 + gn]      = (_Float16)re;
                        o[HW + gm * N320 + gn] = (_Float16)im;
                    } else {
                        C32[b * cB + gm * N320 + gn] = make_float2(re, im);
                    }
                }
    }
}

// ---------------------------------------------------------------------------
// Producers / consumers around the GEMMs (convert to/from f16 planes)
// ---------------------------------------------------------------------------
__global__ void kmask_to_planes(const float2* __restrict__ K, const float* __restrict__ m,
                                _Float16* __restrict__ out) {
    int idx = blockIdx.x * 256 + threadIdx.x;
    if (idx >= SC * HW) return;
    int s  = idx / (C_ * HW);
    int hw = idx % HW;
    float mm = m[s * HW + hw];
    float2 v = K[idx];
    long bp = (long)(idx / HW) * PL2;
    out[bp + hw]      = (_Float16)(v.x * mm);
    out[bp + HW + hw] = (_Float16)(v.y * mm);
}

__global__ void coil_expand_f16(const float2* __restrict__ p, const float2* __restrict__ csm,
                                _Float16* __restrict__ out) {
    int idx = blockIdx.x * 256 + threadIdx.x;
    if (idx >= SC * HW) return;
    int s  = idx / (C_ * HW);
    int c  = (idx / HW) % C_;
    int hw = idx % HW;
    float2 a = p[s * HW + hw];
    float2 b = csm[c * HW + hw];
    long bp = (long)(idx / HW) * PL2;
    out[bp + hw]      = (_Float16)(a.x * b.x - a.y * b.y);
    out[bp + HW + hw] = (_Float16)(a.x * b.y + a.y * b.x);
}

__global__ void coil_combine_f16(const _Float16* __restrict__ z, const float2* __restrict__ csm,
                                 const float2* __restrict__ base, float2* __restrict__ out) {
    int idx = blockIdx.x * 256 + threadIdx.x;
    if (idx >= S_ * HW) return;
    int s  = idx / HW;
    int hw = idx - s * HW;
    float re = 0.f, im = 0.f;
    #pragma unroll 4
    for (int c = 0; c < C_; ++c) {
        long bp = (long)(s * C_ + c) * PL2;
        float zr = (float)z[bp + hw];
        float zi = (float)z[bp + HW + hw];
        float2 cs = csm[c * HW + hw];
        re += cs.x * zr + cs.y * zi;     // conj(cs) * z
        im += cs.x * zi - cs.y * zr;
    }
    float2 bb = base[idx];
    out[idx] = make_float2(re + RHO * bb.x, im + RHO * bb.y);
}

__global__ void to_planes(const float2* __restrict__ x, _Float16* __restrict__ out) {
    int idx = blockIdx.x * 256 + threadIdx.x;
    if (idx >= S_ * HW) return;
    long bp = (long)(idx / HW) * PL2;
    int  hw = idx % HW;
    float2 v = x[idx];
    out[bp + hw]      = (_Float16)v.x;
    out[bp + HW + hw] = (_Float16)v.y;
}

// ---------------------------------------------------------------------------
// CG scalar machinery (deterministic two-stage reductions)
// ---------------------------------------------------------------------------
__global__ __launch_bounds__(256)
void dot_partial(const float* __restrict__ a, const float* __restrict__ b,
                 float* __restrict__ part, int n) {
    __shared__ float sm[256];
    float s = 0.f;
    for (int i = blockIdx.x * 256 + threadIdx.x; i < n; i += 256 * NBLK)
        s += a[i] * b[i];
    sm[threadIdx.x] = s;
    __syncthreads();
    for (int off = 128; off > 0; off >>= 1) {
        if ((int)threadIdx.x < off) sm[threadIdx.x] += sm[threadIdx.x + off];
        __syncthreads();
    }
    if (threadIdx.x == 0) part[blockIdx.x] = sm[0];
}

// scal: [0]=rs [1]=alpha [2]=beta.  mode 0: rs=tot; 1: alpha=rs/(tot+eps);
// 2: beta=tot/(rs+eps), rs=tot
__global__ __launch_bounds__(256)
void reduce_scalar(const float* __restrict__ part, float* __restrict__ scal, int mode) {
    __shared__ float sm[256];
    sm[threadIdx.x] = ((int)threadIdx.x < NBLK) ? part[threadIdx.x] : 0.f;
    __syncthreads();
    for (int off = 128; off > 0; off >>= 1) {
        if ((int)threadIdx.x < off) sm[threadIdx.x] += sm[threadIdx.x + off];
        __syncthreads();
    }
    if (threadIdx.x == 0) {
        float tot = sm[0];
        if (mode == 0)      { scal[0] = tot; }
        else if (mode == 1) { scal[1] = scal[0] / (tot + EPS_); }
        else                { scal[2] = tot / (scal[0] + EPS_); scal[0] = tot; }
    }
}

__global__ void seed_xp(float* __restrict__ x, float* __restrict__ p,
                        const float* __restrict__ r, int n) {
    int i = blockIdx.x * 256 + threadIdx.x;
    if (i < n) { x[i] = 0.f; p[i] = r[i]; }
}

__global__ void update_xr(float* __restrict__ x, float* __restrict__ r,
                          const float* __restrict__ p, const float* __restrict__ ap,
                          const float* __restrict__ scal, int n) {
    int i = blockIdx.x * 256 + threadIdx.x;
    if (i >= n) return;
    float alpha = scal[1];
    x[i] += alpha * p[i];
    r[i] -= alpha * ap[i];
}

__global__ void update_p(float* __restrict__ p, const float* __restrict__ r,
                         const float* __restrict__ scal, int n) {
    int i = blockIdx.x * 256 + threadIdx.x;
    if (i >= n) return;
    float beta = scal[2];
    p[i] = r[i] + beta * p[i];
}

// ---------------------------------------------------------------------------
extern "C" void kernel_launch(void* const* d_in, const int* in_sizes, int n_in,
                              void* d_out, int out_size, void* d_ws, size_t ws_size,
                              hipStream_t stream) {
    (void)in_sizes; (void)n_in; (void)out_size; (void)ws_size;

    const float2* K    = (const float2*)d_in[0];
    const float2* I    = (const float2*)d_in[1];
    const float2* csm  = (const float2*)d_in[2];
    const float*  mask = (const float*) d_in[3];

    // ---- workspace carve-up (~67 MB) ----
    _Float16* Wf16   = (_Float16*)d_ws;
    _Float16* Wi16   = Wf16 + PL2;
    _Float16* bufA16 = Wi16 + PL2;                 // SC * PL2 halfs
    _Float16* bufB16 = bufA16 + (long)SC * PL2;    // holds transposed pass-1 out
    float2*   r      = (float2*)(bufB16 + (long)SC * PL2);
    float2*   p      = r + (long)S_ * HW;
    float2*   x      = p + (long)S_ * HW;
    float2*   ap     = x + (long)S_ * HW;
    float*    part   = (float*)(ap + (long)S_ * HW);
    float*    scal   = part + NBLK;

    const int nF  = S_ * HW * 2;
    const int gF  = (nF + 255) / 256;
    const int gSC = (SC * HW + 255) / 256;
    const int gS  = (S_ * HW + 255) / 256;
    const int gHW = (HW + 255) / 256;

    auto GEMM = [&](const _Float16* A, const _Float16* BT, _Float16* c16, float2* c32,
                    const float* mk, int batch, long aS, long bS, long cS, int sT) {
        cgemm_wmma<<<dim3(N320 / TM, N320 / TN, batch), 128, 0, stream>>>(
            A, BT, c16, c32, mk, aS, bS, cS, sT);
    };

    build_dft_f16<<<gHW, 256, 0, stream>>>(Wf16, Wi16);

    // ---- rhs = EH(kdata) + rho*img -> r ; x=0 ; p=r ; rs=<r,r> ----
    kmask_to_planes<<<gSC, 256, 0, stream>>>(K, mask, bufA16);
    GEMM(bufA16, Wi16, bufB16, nullptr, nullptr, SC, PL2, 0, PL2, 1);  // t^T=((mk)V)^T
    GEMM(Wi16, bufB16, bufA16, nullptr, nullptr, SC, 0, PL2, PL2, 0);  // z = V t
    coil_combine_f16<<<gS, 256, 0, stream>>>(bufA16, csm, I, r);
    seed_xp<<<gF, 256, 0, stream>>>((float*)x, (float*)p, (const float*)r, nF);
    dot_partial<<<NBLK, 256, 0, stream>>>((const float*)r, (const float*)r, part, nF);
    reduce_scalar<<<1, 256, 0, stream>>>(part, scal, 0);

    // ---- 15 CG iterations ----
    for (int it = 0; it < CG_ITERS; ++it) {
        coil_expand_f16<<<gSC, 256, 0, stream>>>(p, csm, bufA16);                // s*p
        GEMM(bufA16, Wf16, bufB16, nullptr, nullptr, SC, PL2, 0, PL2, 1);        // t^T
        GEMM(Wf16, bufB16, bufA16, nullptr, mask,    SC, 0, PL2, PL2, 0);        // m*W t
        GEMM(bufA16, Wi16, bufB16, nullptr, nullptr, SC, PL2, 0, PL2, 1);        // t2^T
        GEMM(Wi16, bufB16, bufA16, nullptr, nullptr, SC, 0, PL2, PL2, 0);        // z
        coil_combine_f16<<<gS, 256, 0, stream>>>(bufA16, csm, p, ap);            // Ap

        dot_partial<<<NBLK, 256, 0, stream>>>((const float*)p, (const float*)ap, part, nF);
        reduce_scalar<<<1, 256, 0, stream>>>(part, scal, 1);
        update_xr<<<gF, 256, 0, stream>>>((float*)x, (float*)r, (const float*)p,
                                          (const float*)ap, scal, nF);
        dot_partial<<<NBLK, 256, 0, stream>>>((const float*)r, (const float*)r, part, nF);
        reduce_scalar<<<1, 256, 0, stream>>>(part, scal, 2);
        update_p<<<gF, 256, 0, stream>>>((float*)p, (const float*)r, scal, nF);
    }

    // ---- out = ifft2c(fft2c(x)), final pass writes f32 straight to d_out ----
    to_planes<<<gS, 256, 0, stream>>>(x, bufA16);
    GEMM(bufA16, Wf16, bufB16, nullptr, nullptr, S_, PL2, 0, PL2, 1);
    GEMM(Wf16, bufB16, bufA16, nullptr, nullptr, S_, 0, PL2, PL2, 0);
    GEMM(bufA16, Wi16, bufB16, nullptr, nullptr, S_, PL2, 0, PL2, 1);
    GEMM(Wi16, bufB16, nullptr, (float2*)d_out, nullptr, S_, 0, PL2, (long)HW, 0);
}
